// SRAttention_79336635891919
// MI455X (gfx1250) — matrix operand order repaired and checked by reading, not compiled, not run on hardware
//
#include <hip/hip_runtime.h>
#include <hip/hip_bf16.h>
#include <cstdint>
#include <cstddef>

// ---------------------------------------------------------------------------
// Problem constants (from the reference)
// ---------------------------------------------------------------------------
#define BB        16          // batch
#define NN        4096        // tokens (H*W)
#define CC        512         // channels
#define HD        64          // head dim
#define NR        256         // reduced tokens N/R^2
#define KSR       8192        // R*R*C
#define LN_EPS    1e-5f

typedef __bf16 bf16;
typedef __attribute__((ext_vector_type(16))) __bf16 v16bf;
typedef __attribute__((ext_vector_type(8)))  float  v8f;
typedef __attribute__((ext_vector_type(4)))  unsigned int u32x4;
typedef __attribute__((ext_vector_type(8)))  int i32x8;
typedef __attribute__((ext_vector_type(4)))  int i32x4;

// TDM builtin availability (device pass only; host pass uses fallback parse)
#if defined(__HIP_DEVICE_COMPILE__) && defined(__has_builtin)
# if __has_builtin(__builtin_amdgcn_tensor_load_to_lds)
#  define HAVE_TDM 1
# endif
#endif
#ifndef HAVE_TDM
# define HAVE_TDM 0
#endif

// ---------------------------------------------------------------------------
// WMMA fragment loaders (contiguous K = k0 + 16*half + e in A and B, 16-bit 16x16x32)
// ---------------------------------------------------------------------------
// A fragment: lane = {m = lane&15, half = lane>>4}; elements e = 0..15 hold
// K = k0 + 16*half + e (contiguous, matching B).
static __device__ inline v16bf a_frag_f32(const float* __restrict__ aRow,
                                          int k0, int half) {
  const float4* ap = (const float4*)(aRow + k0 + 16 * half);
  float4 p0 = ap[0], p1 = ap[1];       // K = k0+16h .. +7
  float4 p2 = ap[2], p3 = ap[3];       // K = k0+16h+8 .. +15
  v16bf a;
  a[0]  = (bf16)p0.x; a[1]  = (bf16)p0.y; a[2]  = (bf16)p0.z; a[3]  = (bf16)p0.w;
  a[4]  = (bf16)p1.x; a[5]  = (bf16)p1.y; a[6]  = (bf16)p1.z; a[7]  = (bf16)p1.w;
  a[8]  = (bf16)p2.x; a[9]  = (bf16)p2.y; a[10] = (bf16)p2.z; a[11] = (bf16)p2.w;
  a[12] = (bf16)p3.x; a[13] = (bf16)p3.y; a[14] = (bf16)p3.z; a[15] = (bf16)p3.w;
  return a;
}

// B fragment (B = W^T, W row-major [N,K]): lane = {n = lane&15, half},
// elements e = 0..15 hold K = k0 + 16*half + e  -> 16 contiguous elements.
static __device__ inline v16bf b_frag_bf16(const bf16* __restrict__ wRow,
                                           int k0, int half) {
  return *(const v16bf*)(wRow + k0 + 16 * half);
}

static __device__ inline v16bf b_frag_f32(const float* __restrict__ wRow,
                                          int k0, int half) {
  const float4* wp = (const float4*)(wRow + k0 + 16 * half);
  float4 q0 = wp[0], q1 = wp[1], q2 = wp[2], q3 = wp[3];
  v16bf v;
  v[0]  = (bf16)q0.x; v[1]  = (bf16)q0.y; v[2]  = (bf16)q0.z; v[3]  = (bf16)q0.w;
  v[4]  = (bf16)q1.x; v[5]  = (bf16)q1.y; v[6]  = (bf16)q1.z; v[7]  = (bf16)q1.w;
  v[8]  = (bf16)q2.x; v[9]  = (bf16)q2.y; v[10] = (bf16)q2.z; v[11] = (bf16)q2.w;
  v[12] = (bf16)q3.x; v[13] = (bf16)q3.y; v[14] = (bf16)q3.z; v[15] = (bf16)q3.w;
  return v;
}

// ---------------------------------------------------------------------------
// Generic batched GEMM:  C[b] = alpha * (A[b] @ W[b]^T) + bias
// (used for all small/medium GEMMs; the big SR GEMM has a TDM kernel below)
// ---------------------------------------------------------------------------
template <bool WBF16>
__global__ __launch_bounds__(256) void gemm_xwt_kernel(
    const float* __restrict__ A, const void* __restrict__ Wv,
    const float* __restrict__ bias, float* __restrict__ C,
    int M, int Nout, int K, int lda, int ldw, int ldc,
    long long sA, long long sW, long long sC, float alpha) {
  const int lane = threadIdx.x & 31;
  const int wib  = threadIdx.x >> 5;
  const long long unit = (long long)blockIdx.x * (blockDim.x >> 5) + wib;
  const int mTiles  = M >> 4;
  const int nGroups = Nout >> 6;
  if (unit >= (long long)mTiles * nGroups) return;   // whole-wave exit
  const int mt = (int)(unit % mTiles);
  const int ng = (int)(unit / mTiles);

  const int m = lane & 15, half = lane >> 4;
  const float* Ab = A + (long long)blockIdx.z * sA;
  float*       Cb = C + (long long)blockIdx.z * sC;
  const float* aRow = Ab + (long long)(mt * 16 + m) * lda;

  const bf16*  wr_b[4];
  const float* wr_f[4];
  if constexpr (WBF16) {
    const bf16* Wb = (const bf16*)Wv + (long long)blockIdx.z * sW;
#pragma unroll
    for (int t = 0; t < 4; ++t)
      wr_b[t] = Wb + (long long)(ng * 64 + t * 16 + m) * ldw;
  } else {
    const float* Wf = (const float*)Wv + (long long)blockIdx.z * sW;
#pragma unroll
    for (int t = 0; t < 4; ++t)
      wr_f[t] = Wf + (long long)(ng * 64 + t * 16 + m) * ldw;
  }

  v8f acc0 = {}, acc1 = {}, acc2 = {}, acc3 = {};
  for (int k0 = 0; k0 < K; k0 += 32) {
    v16bf a = a_frag_f32(aRow, k0, half);
    if (k0 + 64 < K)  // speculative prefetch of next A K-slab
      __builtin_prefetch((const void*)(aRow + k0 + 64), 0, 0);
    v16bf b0, b1, b2, b3;
    if constexpr (WBF16) {
      b0 = b_frag_bf16(wr_b[0], k0, half);
      b1 = b_frag_bf16(wr_b[1], k0, half);
      b2 = b_frag_bf16(wr_b[2], k0, half);
      b3 = b_frag_bf16(wr_b[3], k0, half);
    } else {
      b0 = b_frag_f32(wr_f[0], k0, half);
      b1 = b_frag_f32(wr_f[1], k0, half);
      b2 = b_frag_f32(wr_f[2], k0, half);
      b3 = b_frag_f32(wr_f[3], k0, half);
    }
    acc0 = __builtin_amdgcn_wmma_f32_16x16x32_bf16(false, a, false, b0, (short)0, acc0, false, false);
    acc1 = __builtin_amdgcn_wmma_f32_16x16x32_bf16(false, a, false, b1, (short)0, acc1, false, false);
    acc2 = __builtin_amdgcn_wmma_f32_16x16x32_bf16(false, a, false, b2, (short)0, acc2, false, false);
    acc3 = __builtin_amdgcn_wmma_f32_16x16x32_bf16(false, a, false, b3, (short)0, acc3, false, false);
  }

#pragma unroll
  for (int t = 0; t < 4; ++t) {
    const int ccol = ng * 64 + t * 16 + m;
    const float bv = bias ? bias[ccol] : 0.0f;
    v8f acc = (t == 0) ? acc0 : (t == 1) ? acc1 : (t == 2) ? acc2 : acc3;
#pragma unroll
    for (int r = 0; r < 8; ++r) {
      const int crow = mt * 16 + half * 8 + r;
      Cb[(long long)crow * ldc + ccol] = acc[r] * alpha + bv;
    }
  }
}

// ---------------------------------------------------------------------------
// TDM descriptor issue: load an 8-row x 32-col bf16 slice of W_sr (row-major
// [512, 8192]) into LDS.  2D tile: dim0 = 32 contiguous K elements (2B each),
// dim1 = 8 rows, row stride = 8192 elements.  D# layout per ISA ch.8.
// ---------------------------------------------------------------------------
#if HAVE_TDM
static __device__ inline void tdm_load_slice(const bf16* W, int n0, int k0,
                                             unsigned lds_addr) {
  unsigned long long ga =
      (unsigned long long)(uintptr_t)(W + ((long long)n0 * KSR + k0));
  u32x4 g0;
  g0.x = 1u;                                   // count=1 valid descriptor
  g0.y = lds_addr;                             // LDS byte address
  g0.z = (unsigned)ga;                         // global_addr[31:0]
  g0.w = (unsigned)((ga >> 32) & 0x01FFFFFFu) | (2u << 30);  // addr hi | type=2
  i32x8 g1;
  g1[0] = 0x00010000;                          // wg_mask=0, data_size=1 (2B)
  g1[1] = (int)((KSR & 0xFFFF) << 16);         // tensor_dim0 lo16 (bits 63:48)
  g1[2] = (int)(((KSR >> 16) & 0xFFFF) | ((CC & 0xFFFF) << 16)); // dim0 hi | dim1 lo
  g1[3] = (int)(((CC >> 16) & 0xFFFF) | (32 << 16));             // dim1 hi | tile_dim0=32
  g1[4] = 8;                                   // tile_dim1=8, tile_dim2=0
  g1[5] = KSR;                                 // tensor_dim0_stride lo32
  g1[6] = 0;                                   // stride hi | dim1_stride lo
  g1[7] = 0;
  i32x4 z4 = {0, 0, 0, 0};
#if __clang_major__ >= 23
  i32x8 z8 = {0, 0, 0, 0, 0, 0, 0, 0};
  __builtin_amdgcn_tensor_load_to_lds(g0, g1, z4, z4, z8, 0);
#else
  __builtin_amdgcn_tensor_load_to_lds(g0, g1, z4, z4, 0);
#endif
}
#endif

// ---------------------------------------------------------------------------
// SR GEMM (the 34-GFLOP one, x2):  C = A @ W^T + bias
//   A: fp32 [4096, 8192]   W: bf16 [512, 8192]   C: fp32 [4096, 512]
// Workgroup = 8 waves -> 128x64 macro-tile. The 64x32 bf16 B-panel for each
// K-step is staged in LDS: each wave TDM-loads its own 8-row slice into a
// double buffer (own TENSORcnt wait -> workgroup barrier publishes it), so
// B-matrix traffic is shared 8 ways and moved off the VGPR load path.
// ---------------------------------------------------------------------------
__global__ __launch_bounds__(256) void gemm_sr_kernel(
    const float* __restrict__ A, const bf16* __restrict__ W,
    const float* __restrict__ bias, float* __restrict__ C) {
  __shared__ __align__(64) bf16 panel[2][64 * 32];   // 2 x 4KB

  const int lane = threadIdx.x & 31;
  const int wib  = threadIdx.x >> 5;                 // 0..7
  const int uw   = __builtin_amdgcn_readfirstlane(wib);  // force SGPR-uniform
  const int mb   = blockIdx.x & 31;                  // 32 m-blocks of 128 rows
  const int ng   = blockIdx.x >> 5;                  // 8 n-groups of 64 cols
  const int m = lane & 15, half = lane >> 4;

  const float* aRow = A + (long long)(mb * 128 + wib * 16 + m) * KSR;
  const int n0 = ng * 64;

  const unsigned ldsBase0 = (unsigned)(uintptr_t)(void*)&panel[0][0];
  const unsigned ldsBase1 = (unsigned)(uintptr_t)(void*)&panel[1][0];
  const unsigned sliceOff = (unsigned)(uw * 8 * 32 * 2);  // 8 rows x 64B

#if HAVE_TDM
  tdm_load_slice(W, n0 + uw * 8, 0, ldsBase0 + sliceOff);
  __builtin_amdgcn_s_wait_tensorcnt(0);
#else
  {  // cooperative vector-load staging fallback (same LDS layout)
    const int rloc = uw * 8 + (lane >> 2), chunk = lane & 3;
    const uint4* src = (const uint4*)(W + (long long)(n0 + rloc) * KSR) + chunk;
    *((uint4*)(panel[0] + rloc * 32) + chunk) = *src;
  }
#endif
  __syncthreads();

  v8f acc0 = {}, acc1 = {}, acc2 = {}, acc3 = {};
  const int nSteps = KSR / 32;
  for (int ks = 0; ks < nSteps; ++ks) {
    const int k0  = ks * 32;
    const int cur = ks & 1;
    // kick off next K-step's B-panel into the other buffer
    if (ks + 1 < nSteps) {
#if HAVE_TDM
      tdm_load_slice(W, n0 + uw * 8, k0 + 32,
                     (cur ? ldsBase0 : ldsBase1) + sliceOff);
#else
      const int rloc = uw * 8 + (lane >> 2), chunk = lane & 3;
      const uint4* src =
          (const uint4*)(W + (long long)(n0 + rloc) * KSR + (k0 + 32)) + chunk;
      *((uint4*)(panel[cur ^ 1] + rloc * 32) + chunk) = *src;
#endif
    }

    v16bf a = a_frag_f32(aRow, k0, half);
    if (k0 + 64 < KSR)
      __builtin_prefetch((const void*)(aRow + k0 + 64), 0, 0);

    const bf16* p = &panel[cur][0];
    v16bf b0 = *(const v16bf*)(p + (0 * 16 + m) * 32 + 16 * half);
    v16bf b1 = *(const v16bf*)(p + (1 * 16 + m) * 32 + 16 * half);
    v16bf b2 = *(const v16bf*)(p + (2 * 16 + m) * 32 + 16 * half);
    v16bf b3 = *(const v16bf*)(p + (3 * 16 + m) * 32 + 16 * half);

    acc0 = __builtin_amdgcn_wmma_f32_16x16x32_bf16(false, a, false, b0, (short)0, acc0, false, false);
    acc1 = __builtin_amdgcn_wmma_f32_16x16x32_bf16(false, a, false, b1, (short)0, acc1, false, false);
    acc2 = __builtin_amdgcn_wmma_f32_16x16x32_bf16(false, a, false, b2, (short)0, acc2, false, false);
    acc3 = __builtin_amdgcn_wmma_f32_16x16x32_bf16(false, a, false, b3, (short)0, acc3, false, false);

#if HAVE_TDM
    __builtin_amdgcn_s_wait_tensorcnt(0);   // own slice for next step landed
#endif
    __syncthreads();                        // publish next buffer to all waves
  }

#pragma unroll
  for (int t = 0; t < 4; ++t) {
    const int ccol = n0 + t * 16 + m;
    const float bv = bias[ccol];
    v8f acc = (t == 0) ? acc0 : (t == 1) ? acc1 : (t == 2) ? acc2 : acc3;
#pragma unroll
    for (int r = 0; r < 8; ++r) {
      const int crow = mb * 128 + wib * 16 + half * 8 + r;
      C[(long long)crow * CC + ccol] = acc[r] + bv;
    }
  }
}

// ---------------------------------------------------------------------------
// Weight prep
// ---------------------------------------------------------------------------
__global__ void f32_to_bf16_kernel(const float* __restrict__ in,
                                   bf16* __restrict__ out, int n) {
  int i = blockIdx.x * blockDim.x + threadIdx.x;
  if (i < n) out[i] = (bf16)in[i];
}

// W_eff[o,j] = sum_h W_L2[o, h*64+j]   (tile-by-8 of head output folded in)
__global__ void weff_kernel(const float* __restrict__ WL2,
                            bf16* __restrict__ weff) {
  int i = blockIdx.x * blockDim.x + threadIdx.x;
  if (i >= CC * HD) return;
  int o = i >> 6, j = i & 63;
  float s = 0.f;
#pragma unroll
  for (int h = 0; h < 8; ++h) s += WL2[o * CC + h * HD + j];
  weff[i] = (bf16)s;
}

// ---------------------------------------------------------------------------
// LayerNorm over C=512, one wave per row (population variance, eps=1e-5)
// ---------------------------------------------------------------------------
__global__ __launch_bounds__(256) void ln_kernel(float* __restrict__ y,
                                                 const float* __restrict__ g,
                                                 const float* __restrict__ beta,
                                                 int rows) {
  int wid  = (int)((blockIdx.x * blockDim.x + threadIdx.x) >> 5);
  int lane = threadIdx.x & 31;
  if (wid >= rows) return;
  float* r = y + (long long)wid * CC;
  float v[16], s = 0.f, sq = 0.f;
#pragma unroll
  for (int i = 0; i < 16; ++i) {
    v[i] = r[lane + 32 * i];
    s += v[i]; sq += v[i] * v[i];
  }
#pragma unroll
  for (int off = 16; off >= 1; off >>= 1) {
    s  += __shfl_xor(s, off);
    sq += __shfl_xor(sq, off);
  }
  float mu  = s * (1.0f / CC);
  float var = sq * (1.0f / CC) - mu * mu;
  float rs  = rsqrtf(var + LN_EPS);
#pragma unroll
  for (int i = 0; i < 16; ++i) {
    int idx = lane + 32 * i;
    r[idx] = (v[i] - mu) * rs * g[idx] + beta[idx];
  }
}

// ---------------------------------------------------------------------------
// Column softmax over axis n (size 4096) of scores [B, 4096, 256]
// ---------------------------------------------------------------------------
__global__ __launch_bounds__(256) void colmax_kernel(const float* __restrict__ S,
                                                     float* __restrict__ Pmax) {
  int b = blockIdx.x >> 3, c = blockIdx.x & 7, t = threadIdx.x;
  const float* base = S + ((long long)b * NN + c * 512) * NR + t;
  float mx = -3.4e38f;
  for (int i = 0; i < 512; ++i) mx = fmaxf(mx, base[(long long)i * NR]);
  Pmax[(blockIdx.x << 8) + t] = mx;
}

__global__ __launch_bounds__(256) void exp_partial_kernel(float* __restrict__ S,
                                                          const float* __restrict__ Pmax,
                                                          float* __restrict__ Psum) {
  int b = blockIdx.x >> 3, c = blockIdx.x & 7, t = threadIdx.x;
  float cm = -3.4e38f;
#pragma unroll
  for (int cc = 0; cc < 8; ++cc) cm = fmaxf(cm, Pmax[((b << 3) + cc) * NR + t]);
  float* base = S + ((long long)b * NN + c * 512) * NR + t;
  float s = 0.f;
  for (int i = 0; i < 512; ++i) {
    float e = __expf(base[(long long)i * NR] - cm);
    base[(long long)i * NR] = e;
    s += e;
  }
  Psum[(blockIdx.x << 8) + t] = s;
}

// fold 1/colsum into transposed srv:  srvT[b][d][m] = srv[b][m][d] / S[m]
__global__ __launch_bounds__(256) void svt_kernel(const float* __restrict__ srv,
                                                  const float* __restrict__ Psum,
                                                  float* __restrict__ srvT) {
  int b = blockIdx.x, t = threadIdx.x;  // t = m
  float s = 0.f;
#pragma unroll
  for (int c = 0; c < 8; ++c) s += Psum[((b << 3) + c) * NR + t];
  float inv = 1.0f / s;
  const float* sp = srv + ((long long)b * NR + t) * HD;
  float* dp = srvT + (long long)b * HD * NR + t;
  for (int d = 0; d < HD; ++d) dp[(long long)d * NR] = sp[d] * inv;
}

// ---------------------------------------------------------------------------
// Launch
// ---------------------------------------------------------------------------
extern "C" void kernel_launch(void* const* d_in, const int* in_sizes, int n_in,
                              void* d_out, int out_size, void* d_ws, size_t ws_size,
                              hipStream_t stream) {
  (void)in_sizes; (void)n_in; (void)out_size; (void)ws_size;
  const float* query = (const float*)d_in[0];
  const float* key_  = (const float*)d_in[1];
  const float* value = (const float*)d_in[2];
  const float* W_L   = (const float*)d_in[3];
  const float* b_L   = (const float*)d_in[4];
  const float* W_sr  = (const float*)d_in[5];
  const float* b_sr  = (const float*)d_in[6];
  const float* ln_g  = (const float*)d_in[7];
  const float* ln_b  = (const float*)d_in[8];
  const float* W_L2  = (const float*)d_in[9];
  const float* b_L2  = (const float*)d_in[10];
  float* out = (float*)d_out;
  char*  ws  = (char*)d_ws;

  // workspace layout (bytes, 256B aligned); total ~103 MB
  const size_t OFF_WLB   = 0;                                        // bf16 [64,512]
  const size_t OFF_WSRB  = OFF_WLB   + (size_t)HD * CC * 2;          // bf16 [512,8192]
  const size_t OFF_WEFFB = OFF_WSRB  + (size_t)CC * KSR * 2;         // bf16 [512,64]
  const size_t OFF_QI    = OFF_WEFFB + (size_t)CC * HD * 2;          // f32 [B*N,64] (reused as Ai)
  const size_t OFF_Y     = OFF_QI    + (size_t)BB * NN * HD * 4;     // f32 [B*NR,512]
  const size_t OFF_SRK   = OFF_Y     + (size_t)BB * NR * CC * 4;     // f32 [B*NR,64]
  const size_t OFF_SRV   = OFF_SRK   + (size_t)BB * NR * HD * 4;     // f32 [B*NR,64]
  const size_t OFF_SRVT  = OFF_SRV   + (size_t)BB * NR * HD * 4;     // f32 [B,64,256]
  const size_t OFF_PMAX  = OFF_SRVT  + (size_t)BB * HD * NR * 4;     // f32 [B*8*256]
  const size_t OFF_PSUM  = OFF_PMAX  + (size_t)BB * 8 * NR * 4;
  const size_t OFF_SC    = OFF_PSUM  + (size_t)BB * 8 * NR * 4;      // f32 [B,4096,256]

  bf16*  WLb   = (bf16*)(ws + OFF_WLB);
  bf16*  WSRb  = (bf16*)(ws + OFF_WSRB);
  bf16*  WEFFb = (bf16*)(ws + OFF_WEFFB);
  float* qi    = (float*)(ws + OFF_QI);
  float* y     = (float*)(ws + OFF_Y);
  float* srk   = (float*)(ws + OFF_SRK);
  float* srv   = (float*)(ws + OFF_SRV);
  float* srvT  = (float*)(ws + OFF_SRVT);
  float* Pmax  = (float*)(ws + OFF_PMAX);
  float* Psum  = (float*)(ws + OFF_PSUM);
  float* sc    = (float*)(ws + OFF_SC);

  const unsigned T = 256;
  // --- weight prep ---
  f32_to_bf16_kernel<<<(HD * CC + T - 1) / T, T, 0, stream>>>(W_L, WLb, HD * CC);
  f32_to_bf16_kernel<<<(CC * KSR + T - 1) / T, T, 0, stream>>>(W_sr, WSRb, CC * KSR);
  weff_kernel<<<(CC * HD + T - 1) / T, T, 0, stream>>>(W_L2, WEFFb);

  auto gblk = [](long long mTiles, long long nGroups) {
    return (unsigned)((mTiles * nGroups + 7) / 8);
  };

  // --- qi = query @ W_L^T + b_L : [B*N,64]
  gemm_xwt_kernel<true><<<dim3(gblk(4096, 1)), T, 0, stream>>>(
      query, WLb, b_L, qi, BB * NN, HD, CC, CC, CC, HD, 0, 0, 0, 1.0f);

  // --- SR(key): y = Xr @ W_sr^T + b_sr (TDM kernel) ; LN ; srk = y @ W_L^T + b_L
  gemm_sr_kernel<<<dim3(256), T, 0, stream>>>(key_, WSRb, b_sr, y);
  ln_kernel<<<(BB * NR * 32 + T - 1) / T, T, 0, stream>>>(y, ln_g, ln_b, BB * NR);
  gemm_xwt_kernel<true><<<dim3(gblk(256, 1)), T, 0, stream>>>(
      y, WLb, b_L, srk, BB * NR, HD, CC, CC, CC, HD, 0, 0, 0, 1.0f);

  // --- SR(value) -> srv
  gemm_sr_kernel<<<dim3(256), T, 0, stream>>>(value, WSRb, b_sr, y);
  ln_kernel<<<(BB * NR * 32 + T - 1) / T, T, 0, stream>>>(y, ln_g, ln_b, BB * NR);
  gemm_xwt_kernel<true><<<dim3(gblk(256, 1)), T, 0, stream>>>(
      y, WLb, b_L, srv, BB * NR, HD, CC, CC, CC, HD, 0, 0, 0, 1.0f);

  // --- scores[b] = (qi[b] @ srk[b]^T) / 8 : [B,4096,256]  (batched, fp32 W)
  gemm_xwt_kernel<false><<<dim3(gblk(256, 4), 1, BB), T, 0, stream>>>(
      qi, srk, nullptr, sc, NN, NR, HD, HD, HD, NR,
      (long long)NN * HD, (long long)NR * HD, (long long)NN * NR, 0.125f);

  // --- softmax over axis n (query axis) per column m, fold 1/sum into srv^T
  colmax_kernel<<<dim3(BB * 8), T, 0, stream>>>(sc, Pmax);
  exp_partial_kernel<<<dim3(BB * 8), T, 0, stream>>>(sc, Pmax, Psum);
  svt_kernel<<<dim3(BB), T, 0, stream>>>(srv, Psum, srvT);

  // --- Ai[b] = E[b] @ srvT[b]^T : [B,4096,64]  (reuse qi buffer)
  gemm_xwt_kernel<false><<<dim3(gblk(256, 1), 1, BB), T, 0, stream>>>(
      sc, srvT, nullptr, qi, NN, HD, NR, NR, NR, HD,
      (long long)NN * NR, (long long)HD * NR, (long long)NN * HD, 1.0f);

  // --- out = Ai @ W_eff^T + b_L2 : [B*N,512]
  gemm_xwt_kernel<true><<<dim3(gblk(4096, 8)), T, 0, stream>>>(
      qi, WEFFb, b_L2, out, BB * NN, CC, HD, HD, HD, CC, 0, 0, 0, 1.0f);
}
